// MHAttention_71433896067706
// MI455X (gfx1250) — matrix operand order repaired and checked by reading, not compiled, not run on hardware
//
#include <hip/hip_runtime.h>
#include <hip/hip_bf16.h>

// ---------------- problem constants (from reference) ----------------
constexpr int Nseq = 2048;   // sequence length
constexpr int Cch  = 512;    // channels
constexpr int Dh   = 64;     // head dim
constexpr int Kr   = 128;    // low rank
constexpr int NH   = 4;      // heads
constexpr int HD   = NH * Dh;        // 256
constexpr int NPROJ= 3 * HD;         // 768 fused q/k/v rows

// ---------------- CDNA5 WMMA plumbing ----------------
typedef __bf16 bf16;
typedef __attribute__((ext_vector_type(16))) __bf16 v16bf;
typedef __attribute__((ext_vector_type(8)))  float  v8f;

#define DEV static __device__ __forceinline__

constexpr int LDSTR = 40;   // 32 bf16 payload + 8 pad (80B row stride, 16B aligned, bank-friendly)

union FragU { v16bf v; uint4 q[2]; };

// A fragment (16x32 bf16, row-major tile in LDS at `lds`, row stride LDSTR):
// lanes 0-15: M=lane, K 0..7 & 16..23 ; lanes 16-31: M=lane-16, K 8..15 & 24..31
DEV v16bf ldsA_frag(const bf16* lds, int lane) {
    FragU u;
    const int row = lane & 15;
    const int kh  = (lane >> 4) * 8;
    u.q[0] = *(const uint4*)(lds + row * LDSTR + kh);
    u.q[1] = *(const uint4*)(lds + row * LDSTR + kh + 16);
    return u.v;
}

// B fragment (32x16 bf16). Tile stored [n][k] row-major in LDS:
// lanes 0-15: N=lane, K 0..7 & 16..23 ; lanes 16-31: N=lane-16, K 8..15 & 24..31
DEV v16bf ldsB_frag(const bf16* lds, int lane) {
    FragU u;
    const int col = lane & 15;
    const int kh  = (lane >> 4) * 8;
    u.q[0] = *(const uint4*)(lds + col * LDSTR + kh);
    u.q[1] = *(const uint4*)(lds + col * LDSTR + kh + 16);
    return u.v;
}

DEV v8f wmma_bf16(v16bf a, v16bf b, v8f c) {
    return __builtin_amdgcn_wmma_f32_16x16x32_bf16(false, a, false, b, (short)0, c, false, false);
}

DEV v8f v8f_zero() { v8f z = {0.f,0.f,0.f,0.f,0.f,0.f,0.f,0.f}; return z; }

// ---------------- CDNA5 async global->LDS copy (ASYNCcnt path) ----------------
#if defined(__has_builtin)
#if __has_builtin(__builtin_amdgcn_global_load_async_to_lds_b128)
#define HAVE_ASYNC_LDS 1
#endif
#endif
#ifndef HAVE_ASYNC_LDS
#define HAVE_ASYNC_LDS 0
#endif

// b128 flavor takes pointers to int4 vectors in addrspace(1) (global) / addrspace(3) (LDS)
typedef int v4i __attribute__((ext_vector_type(4)));
typedef __attribute__((address_space(1))) v4i g_v4i;
typedef __attribute__((address_space(3))) v4i l_v4i;

// copy 16 bytes (8 bf16) from global to LDS
DEV void copy16(const bf16* gsrc, bf16* ldst) {
#if HAVE_ASYNC_LDS
    __builtin_amdgcn_global_load_async_to_lds_b128((g_v4i*)gsrc, (l_v4i*)ldst, 0, 0);
#else
    *(uint4*)ldst = *(const uint4*)gsrc;
#endif
}

DEV void async_drain() {
#if HAVE_ASYNC_LDS
    asm volatile("s_wait_asynccnt 0x0" ::: "memory");
#endif
}

// ---------------- prep kernels ----------------

// fp32 -> bf16 conversions: x, Wo, and EF^T
__global__ __launch_bounds__(256) void prep_convert(
    const float* __restrict__ x,  bf16* __restrict__ xb,
    const float* __restrict__ Wo, bf16* __restrict__ Wob,
    const float* __restrict__ EF, bf16* __restrict__ EFt)
{
    const int id = blockIdx.x * 256 + threadIdx.x;          // grid covers Nseq*Cch exactly
    if (id < Nseq * Cch) xb[id]  = (bf16)x[id];
    if (id < Cch * HD)   Wob[id] = (bf16)Wo[id];
    if (id < Kr * Nseq) {                                   // EFt[k][n] = EF[n][k]
        const int kk = id >> 11, n = id & (Nseq - 1);
        EFt[id] = (bf16)EF[n * Kr + kk];
    }
}

// Wall[j, c] = sum_d Wp_h[h][e][d] * Wp_in[d][c], j = proj*256 + h*64 + e  (768 x 512)
__global__ __launch_bounds__(256) void prep_fuse(
    const float* __restrict__ Wq_in, const float* __restrict__ Wk_in, const float* __restrict__ Wv_in,
    const float* __restrict__ Wq_h,  const float* __restrict__ Wk_h,  const float* __restrict__ Wv_h,
    bf16* __restrict__ Wall)
{
    const int id = blockIdx.x * 256 + threadIdx.x;          // grid covers 768*512 exactly
    const int cc = id & (Cch - 1);
    const int j  = id >> 9;
    const int proj = j >> 8, h = (j >> 6) & 3, e = j & 63;
    const float* Win = (proj == 0) ? Wq_in : (proj == 1) ? Wk_in : Wv_in;
    const float* Wh  = ((proj == 0) ? Wq_h : (proj == 1) ? Wk_h : Wv_h) + (h * Dh + e) * Dh;
    float s = 0.f;
    for (int d = 0; d < Dh; ++d) s += Wh[d] * Win[d * Cch + cc];
    Wall[id] = (bf16)s;
}

// ---------------- GEMM 1: [2048 x 768] = x_bf16 [2048x512] @ Wall^T ----------------
// Double-buffered LDS, async global->LDS, one barrier per K-step.
// Epilogue: proj0 -> QW fp32 [h][n][c] (+bq) ; proj1 -> KWt bf16 [h][c][n] (+bk) ; proj2 -> VWt (+bv)
__global__ __launch_bounds__(128) void gemm_qkv(
    const bf16* __restrict__ X, const bf16* __restrict__ Wall,
    const float* __restrict__ bq, const float* __restrict__ bk, const float* __restrict__ bv,
    float* __restrict__ QW, bf16* __restrict__ KWt, bf16* __restrict__ VWt)
{
    __shared__ bf16 sA[2][64 * LDSTR];
    __shared__ bf16 sB[2][64 * LDSTR];
    const int t = threadIdx.x, lane = t & 31, wv = t >> 5;
    const int mBase = blockIdx.x * 64, nBase = blockIdx.y * 64;
    const int row = t >> 1, seg = (t & 1) * 16;

    v8f acc[4]; for (int i = 0; i < 4; ++i) acc[i] = v8f_zero();

    auto issue = [&](int kb, int buf) {
        const bf16* gA = X    + (size_t)(mBase + row) * Cch + kb + seg;
        const bf16* gB = Wall + (size_t)(nBase + row) * Cch + kb + seg;
        copy16(gA,     &sA[buf][row * LDSTR + seg]);
        copy16(gA + 8, &sA[buf][row * LDSTR + seg + 8]);
        copy16(gB,     &sB[buf][row * LDSTR + seg]);
        copy16(gB + 8, &sB[buf][row * LDSTR + seg + 8]);
    };

    issue(0, 0);
    async_drain();
    __syncthreads();

    int buf = 0;
    for (int kb = 0; kb < Cch; kb += 32) {
        if (kb + 32 < Cch) issue(kb + 32, buf ^ 1);

        // gather all fragments first, then burst the WMMAs
        v16bf a  = ldsA_frag(&sA[buf][(wv * 16) * LDSTR], lane);
        v16bf b0 = ldsB_frag(&sB[buf][ 0 * LDSTR], lane);
        v16bf b1 = ldsB_frag(&sB[buf][16 * LDSTR], lane);
        v16bf b2 = ldsB_frag(&sB[buf][32 * LDSTR], lane);
        v16bf b3 = ldsB_frag(&sB[buf][48 * LDSTR], lane);
        acc[0] = wmma_bf16(a, b0, acc[0]);
        acc[1] = wmma_bf16(a, b1, acc[1]);
        acc[2] = wmma_bf16(a, b2, acc[2]);
        acc[3] = wmma_bf16(a, b3, acc[3]);

        async_drain();
        __syncthreads();
        buf ^= 1;
    }

    for (int f = 0; f < 4; ++f) {
        const int ng   = nBase + f * 16 + (lane & 15);
        const int proj = ng >> 8;
        const int h    = (ng >> 6) & 3;
        const int c    = ng & 63;
        const float bias = ((proj == 0) ? bq : (proj == 1) ? bk : bv)[h * Dh + c];
        for (int r = 0; r < 8; ++r) {
            const int m = mBase + wv * 16 + (lane >> 4) * 8 + r;
            const float val = acc[f][r] + bias;
            if (proj == 0)      QW [((size_t)h * Nseq + m) * Dh + c]  = val;
            else if (proj == 1) KWt[((size_t)h * Dh + c) * Nseq + m]  = (bf16)val;
            else                VWt[((size_t)h * Dh + c) * Nseq + m]  = (bf16)val;
        }
    }
}

// ---------------- GEMM 2: per (head, K/V): [128 x 64] = EFt [128x2048] @ (KWt[h])^T ----------------
__global__ __launch_bounds__(128) void gemm_ef(
    const bf16* __restrict__ EFt, const bf16* __restrict__ KWt, const bf16* __restrict__ VWt,
    float* __restrict__ KWp, float* __restrict__ VWp)
{
    __shared__ bf16 sA[2][128 * LDSTR];
    __shared__ bf16 sB[2][ 64 * LDSTR];
    const int t = threadIdx.x, lane = t & 31, wv = t >> 5;
    const int h = blockIdx.x;
    const bf16* Bsrc = ((blockIdx.y == 0) ? KWt : VWt) + (size_t)h * Dh * Nseq;
    float*      Out  = ((blockIdx.y == 0) ? KWp : VWp) + (size_t)h * Kr * Dh;
    const int rowB = t >> 1, segB = (t & 1) * 16;

    v8f acc[2][4];
    for (int i = 0; i < 2; ++i) for (int f = 0; f < 4; ++f) acc[i][f] = v8f_zero();

    auto issue = [&](int kb, int buf) {
        // A: 128 rows x 32 cols, one row per thread (4 x 16B)
        const bf16* gA = EFt + (size_t)t * Nseq + kb;
        bf16* dA = &sA[buf][t * LDSTR];
        copy16(gA,      dA);
        copy16(gA + 8,  dA + 8);
        copy16(gA + 16, dA + 16);
        copy16(gA + 24, dA + 24);
        // B: 64 rows x 32 cols, half row per thread
        const bf16* gB = Bsrc + (size_t)rowB * Nseq + kb + segB;
        copy16(gB,     &sB[buf][rowB * LDSTR + segB]);
        copy16(gB + 8, &sB[buf][rowB * LDSTR + segB + 8]);
    };

    issue(0, 0);
    async_drain();
    __syncthreads();

    int buf = 0;
    for (int kb = 0; kb < Nseq; kb += 32) {
        if (kb + 32 < Nseq) issue(kb + 32, buf ^ 1);

        v16bf a0 = ldsA_frag(&sA[buf][(wv * 32)      * LDSTR], lane);
        v16bf a1 = ldsA_frag(&sA[buf][(wv * 32 + 16) * LDSTR], lane);
        v16bf b0 = ldsB_frag(&sB[buf][ 0 * LDSTR], lane);
        v16bf b1 = ldsB_frag(&sB[buf][16 * LDSTR], lane);
        v16bf b2 = ldsB_frag(&sB[buf][32 * LDSTR], lane);
        v16bf b3 = ldsB_frag(&sB[buf][48 * LDSTR], lane);
        acc[0][0] = wmma_bf16(a0, b0, acc[0][0]);
        acc[0][1] = wmma_bf16(a0, b1, acc[0][1]);
        acc[0][2] = wmma_bf16(a0, b2, acc[0][2]);
        acc[0][3] = wmma_bf16(a0, b3, acc[0][3]);
        acc[1][0] = wmma_bf16(a1, b0, acc[1][0]);
        acc[1][1] = wmma_bf16(a1, b1, acc[1][1]);
        acc[1][2] = wmma_bf16(a1, b2, acc[1][2]);
        acc[1][3] = wmma_bf16(a1, b3, acc[1][3]);

        async_drain();
        __syncthreads();
        buf ^= 1;
    }

    for (int i = 0; i < 2; ++i)
        for (int f = 0; f < 4; ++f) {
            const int c = f * 16 + (lane & 15);
            for (int r = 0; r < 8; ++r) {
                const int m = wv * 32 + i * 16 + (lane >> 4) * 8 + r;
                Out[m * Dh + c] = acc[i][f][r];
            }
        }
}

// ---------------- channel-softmax attention: one wave32 per (h, n), 2 channels/lane ----------------
__global__ __launch_bounds__(256) void attn_softmax(
    const float* __restrict__ QW, const float* __restrict__ KWp,
    const float* __restrict__ VWp, bf16* __restrict__ Att)
{
    const int lane = threadIdx.x & 31, wv = threadIdx.x >> 5;
    const int row  = blockIdx.x * 8 + wv;       // 0 .. H*N-1
    const int h    = row >> 11;                 // /2048
    const int n    = row & (Nseq - 1);
    const float scale = 0.125f;                 // 1/sqrt(D)

    const float q0 = QW[((size_t)h * Nseq + n) * Dh + lane];
    const float q1 = QW[((size_t)h * Nseq + n) * Dh + lane + 32];
    const float* kp = KWp + (size_t)h * Kr * Dh;
    const float* vp = VWp + (size_t)h * Kr * Dh;

    float acc0 = 0.f, acc1 = 0.f;
    for (int k = 0; k < Kr; ++k) {
        const float l0 = q0 * kp[k * Dh + lane]      * scale;
        const float l1 = q1 * kp[k * Dh + lane + 32] * scale;
        float m = fmaxf(l0, l1);
        for (int off = 16; off; off >>= 1) m = fmaxf(m, __shfl_xor(m, off, 32));
        const float e0 = __expf(l0 - m), e1 = __expf(l1 - m);
        float s = e0 + e1;
        for (int off = 16; off; off >>= 1) s += __shfl_xor(s, off, 32);
        const float rs = 1.0f / s;
        acc0 += e0 * rs * vp[k * Dh + lane];
        acc1 += e1 * rs * vp[k * Dh + lane + 32];
    }
    Att[(size_t)n * HD + h * Dh + lane]      = (bf16)acc0;
    Att[(size_t)n * HD + h * Dh + lane + 32] = (bf16)acc1;
}

// ---------------- GEMM 3: d_out [2048 x 512] = Att [2048x256] @ Wo^T + bo ----------------
__global__ __launch_bounds__(128) void gemm_out(
    const bf16* __restrict__ Att, const bf16* __restrict__ Wob,
    const float* __restrict__ bo, float* __restrict__ Y)
{
    __shared__ bf16 sA[2][64 * LDSTR];
    __shared__ bf16 sB[2][64 * LDSTR];
    const int t = threadIdx.x, lane = t & 31, wv = t >> 5;
    const int mBase = blockIdx.x * 64, nBase = blockIdx.y * 64;
    const int row = t >> 1, seg = (t & 1) * 16;

    v8f acc[4]; for (int i = 0; i < 4; ++i) acc[i] = v8f_zero();

    auto issue = [&](int kb, int buf) {
        const bf16* gA = Att + (size_t)(mBase + row) * HD + kb + seg;
        const bf16* gB = Wob + (size_t)(nBase + row) * HD + kb + seg;
        copy16(gA,     &sA[buf][row * LDSTR + seg]);
        copy16(gA + 8, &sA[buf][row * LDSTR + seg + 8]);
        copy16(gB,     &sB[buf][row * LDSTR + seg]);
        copy16(gB + 8, &sB[buf][row * LDSTR + seg + 8]);
    };

    issue(0, 0);
    async_drain();
    __syncthreads();

    int buf = 0;
    for (int kb = 0; kb < HD; kb += 32) {
        if (kb + 32 < HD) issue(kb + 32, buf ^ 1);

        v16bf a  = ldsA_frag(&sA[buf][(wv * 16) * LDSTR], lane);
        v16bf b0 = ldsB_frag(&sB[buf][ 0 * LDSTR], lane);
        v16bf b1 = ldsB_frag(&sB[buf][16 * LDSTR], lane);
        v16bf b2 = ldsB_frag(&sB[buf][32 * LDSTR], lane);
        v16bf b3 = ldsB_frag(&sB[buf][48 * LDSTR], lane);
        acc[0] = wmma_bf16(a, b0, acc[0]);
        acc[1] = wmma_bf16(a, b1, acc[1]);
        acc[2] = wmma_bf16(a, b2, acc[2]);
        acc[3] = wmma_bf16(a, b3, acc[3]);

        async_drain();
        __syncthreads();
        buf ^= 1;
    }

    for (int f = 0; f < 4; ++f) {
        const int n = nBase + f * 16 + (lane & 15);
        const float bias = bo[n];
        for (int r = 0; r < 8; ++r) {
            const int m = mBase + wv * 16 + (lane >> 4) * 8 + r;
            Y[(size_t)m * Cch + n] = acc[f][r] + bias;
        }
    }
}

// ---------------- host launcher ----------------
extern "C" void kernel_launch(void* const* d_in, const int* in_sizes, int n_in,
                              void* d_out, int out_size, void* d_ws, size_t ws_size,
                              hipStream_t stream) {
    const float* x     = (const float*)d_in[0];
    const float* Wq_in = (const float*)d_in[1];
    const float* Wk_in = (const float*)d_in[2];
    const float* Wv_in = (const float*)d_in[3];
    const float* Wq_h  = (const float*)d_in[4];
    const float* bq_h  = (const float*)d_in[5];
    const float* Wk_h  = (const float*)d_in[6];
    const float* bk_h  = (const float*)d_in[7];
    const float* Wv_h  = (const float*)d_in[8];
    const float* bv_h  = (const float*)d_in[9];
    const float* Wo    = (const float*)d_in[10];
    const float* bo    = (const float*)d_in[11];
    const float* EF    = (const float*)d_in[12];
    float* out = (float*)d_out;

    // workspace carve-up (256B aligned slabs)
    char*  w   = (char*)d_ws;
    size_t off = 0;
    auto carve = [&](size_t bytes) -> void* {
        void* p = w + off;
        off = (off + bytes + 255) & ~(size_t)255;
        return p;
    };
    bf16*  xb   = (bf16*) carve((size_t)Nseq * Cch * 2);     // x in bf16
    bf16*  Wall = (bf16*) carve((size_t)NPROJ * Cch * 2);    // fused qkv weights
    bf16*  EFt  = (bf16*) carve((size_t)Kr * Nseq * 2);      // EF transposed
    bf16*  Wob  = (bf16*) carve((size_t)Cch * HD * 2);       // Wo in bf16
    float* QW   = (float*)carve((size_t)NH * Nseq * Dh * 4); // Q per head, fp32
    bf16*  KWt  = (bf16*) carve((size_t)NH * Dh * Nseq * 2); // K per head, transposed [h][c][n]
    bf16*  VWt  = (bf16*) carve((size_t)NH * Dh * Nseq * 2); // V per head, transposed
    float* KWp  = (float*)carve((size_t)NH * Kr * Dh * 4);   // EF-projected K
    float* VWp  = (float*)carve((size_t)NH * Kr * Dh * 4);   // EF-projected V
    bf16*  Att  = (bf16*) carve((size_t)Nseq * HD * 2);      // concat-head attention out

    prep_convert<<<(Nseq * Cch) / 256, 256, 0, stream>>>(x, xb, Wo, Wob, EF, EFt);
    prep_fuse   <<<(NPROJ * Cch) / 256, 256, 0, stream>>>(Wq_in, Wk_in, Wv_in,
                                                          Wq_h, Wk_h, Wv_h, Wall);
    gemm_qkv    <<<dim3(Nseq / 64, NPROJ / 64), 128, 0, stream>>>(xb, Wall, bq_h, bk_h, bv_h,
                                                                  QW, KWt, VWt);
    gemm_ef     <<<dim3(NH, 2), 128, 0, stream>>>(EFt, KWt, VWt, KWp, VWp);
    attn_softmax<<<(NH * Nseq) / 8, 256, 0, stream>>>(QW, KWp, VWp, Att);
    gemm_out    <<<dim3(Nseq / 64, Cch / 64), 128, 0, stream>>>(Att, Wob, bo, out);
}
